// CausalSelfAttention_87471303950371
// MI455X (gfx1250) — matrix operand order repaired and checked by reading, not compiled, not run on hardware
//
#include <hip/hip_runtime.h>
#include <hip/hip_bf16.h>

typedef __attribute__((ext_vector_type(16))) _Float16 v16h;
typedef __attribute__((ext_vector_type(8)))  _Float16 v8h;
typedef __attribute__((ext_vector_type(8)))  float    v8f;
typedef __attribute__((ext_vector_type(4)))  unsigned int v4u;
typedef __attribute__((ext_vector_type(8)))  int      v8i;
typedef __attribute__((ext_vector_type(4)))  int      v4i;

// ---------------------------------------------------------------------------
// WMMA helper (gfx1250: V_WMMA_F32_16X16X32_F16, D = A*B + C, fp32 accum)
// ---------------------------------------------------------------------------
__device__ __forceinline__ v8f wmma_f16(v16h a, v16h b, v8f c) {
  return __builtin_amdgcn_wmma_f32_16x16x32_f16(
      false, a, false, b, (short)0, c, false, false);
}

// A-fragment (16x32 f16): lane<16 -> row=lane, K = kbase+{0..7, 16..23}
//                         lane>=16 -> row=lane-16, K = kbase+{8..15, 24..31}
__device__ __forceinline__ v16h load_a_frag(const _Float16* p0, int row, int ld,
                                            int kbase) {
  int lane = threadIdx.x & 31;
  int r = row + (lane & 15);
  int k1 = kbase + ((lane & 16) ? 8 : 0);
  const _Float16* p = p0 + (size_t)r * ld + k1;
  v8h lo = *(const v8h*)(p);
  v8h hi = *(const v8h*)(p + 16);
  v16h f;
#pragma unroll
  for (int i = 0; i < 8; ++i) { f[i] = lo[i]; f[i + 8] = hi[i]; }
  return f;
}

// B-fragment: lane<16 -> col=lane, K=kbase+{0..7, 16..23} ; lane>=16 -> col=lane-16,
// K=kbase+{8..15, 24..31}. Source must be [col][k] contiguous in k.
__device__ __forceinline__ v16h load_b_frag(const _Float16* p0, int col, int ld,
                                            int kbase) {
  int lane = threadIdx.x & 31;
  int c = col + (lane & 15);
  int k1 = kbase + ((lane & 16) ? 8 : 0);
  const _Float16* p = p0 + (size_t)c * ld + k1;
  v8h lo = *(const v8h*)(p);
  v8h hi = *(const v8h*)(p + 16);
  v16h f;
#pragma unroll
  for (int i = 0; i < 8; ++i) { f[i] = lo[i]; f[i + 8] = hi[i]; }
  return f;
}

// ---------------------------------------------------------------------------
// Tensor Data Mover: 2D tile load (global f16 -> LDS), D# per ISA ch.8.
// tile_dim0 elements contiguous per row (x), tile_dim1 rows (y), row stride
// stride0 (all in f16 elements). LDS receives the tile densely row-major.
// This toolchain's builtin takes 6 args (extra int32x8 before cpol).
// ---------------------------------------------------------------------------
__device__ __forceinline__ void tdm_load_2d_f16(
    unsigned lds_off, const _Float16* gsrc, unsigned tensor_d0,
    unsigned tensor_d1, unsigned stride0, unsigned tile_d0, unsigned tile_d1) {
  unsigned long long ga = (unsigned long long)(size_t)gsrc;
  v4u g0;
  g0[0] = 1u;                                  // count=1, user-mode load
  g0[1] = lds_off;                             // LDS byte address
  g0[2] = (unsigned)(ga & 0xFFFFFFFFu);        // global_addr[31:0]
  g0[3] = (unsigned)((ga >> 32) & 0x1FFFFFFu)  // global_addr[56:32]
          | (2u << 30);                        // type = 2 ("image")
  v8i g1;
  g1[0] = (int)(1u << 16);                     // wg_mask=0, data_size=1 (2B)
  g1[1] = (int)((tensor_d0 & 0xFFFFu) << 16);  // tensor_dim0[15:0]
  g1[2] = (int)(((tensor_d0 >> 16) & 0xFFFFu) | ((tensor_d1 & 0xFFFFu) << 16));
  g1[3] = (int)(((tensor_d1 >> 16) & 0xFFFFu) | ((tile_d0 & 0xFFFFu) << 16));
  g1[4] = (int)(tile_d1 & 0xFFFFu);            // tile_dim1; tile_dim2=0
  g1[5] = (int)stride0;                        // tensor_dim0_stride[31:0]
  g1[6] = 0;                                   // stride0[47:32]=0, stride1=0
  g1[7] = 0;
  v4i g2 = {0, 0, 0, 0};                       // dims >2 unused (2D tensor)
  v4i g3 = {0, 0, 0, 0};
  v8i g4 = {0, 0, 0, 0, 0, 0, 0, 0};           // unused extra group (zeroed)
  __builtin_amdgcn_tensor_load_to_lds(g0, g1, g2, g3, g4, 0);
}

// ---------------------------------------------------------------------------
// f16 WMMA GEMM with TDM double-buffered staging and fused bias.
//   A : [M,K] f16 row-major, Bt : [N,K] f16 (weights pre-transposed)
//   C : [M,N] fp32,  C = A * Bt^T + bias
// BM=BN=128, BK=32; 256 threads = 8 waves (4x2), wave tile 32x64.
// Wave 0 drives the TDM pipeline; everyone syncs on workgroup barriers.
// ---------------------------------------------------------------------------
#define GBM 128
#define GBN 128
#define GBK 32

__global__ __launch_bounds__(256) void gemm_tdm_wmma(
    const _Float16* __restrict__ A, const _Float16* __restrict__ Bt,
    const float* __restrict__ bias, float* __restrict__ C,
    int M, int N, int K) {
  __shared__ __align__(16) _Float16 As[2][GBM * GBK];
  __shared__ __align__(16) _Float16 Bs[2][GBN * GBK];

  const int tid = threadIdx.x;
  const int bm = blockIdx.y * GBM;
  const int bn = blockIdx.x * GBN;
  const int wid = tid >> 5;
  const int wm = (wid >> 1) * 32;
  const int wn = (wid & 1) * 64;
  const bool issuer = (wid == 0);
  const int nk = K / GBK;

  v8f acc[2][4];
#pragma unroll
  for (int mi = 0; mi < 2; ++mi)
#pragma unroll
    for (int ni = 0; ni < 4; ++ni) acc[mi][ni] = {};

  // Prologue: DMA tile 0 into buffer 0.
  if (issuer) {
    tdm_load_2d_f16((unsigned)(size_t)&As[0][0], A + (size_t)bm * K,
                    (unsigned)K, (unsigned)M, (unsigned)K, GBK, GBM);
    tdm_load_2d_f16((unsigned)(size_t)&Bs[0][0], Bt + (size_t)bn * K,
                    (unsigned)K, (unsigned)N, (unsigned)K, GBK, GBN);
  }

  for (int kt = 0; kt < nk; ++kt) {
    if (issuer) __builtin_amdgcn_s_wait_tensorcnt(0);
    __syncthreads();  // tile kt visible; previous compute on buf[(kt+1)&1] done
    const int cur = kt & 1;
    if (issuer && (kt + 1) < nk) {
      const int k0 = (kt + 1) * GBK;
      tdm_load_2d_f16((unsigned)(size_t)&As[cur ^ 1][0],
                      A + (size_t)bm * K + k0, (unsigned)K, (unsigned)M,
                      (unsigned)K, GBK, GBM);
      tdm_load_2d_f16((unsigned)(size_t)&Bs[cur ^ 1][0],
                      Bt + (size_t)bn * K + k0, (unsigned)K, (unsigned)N,
                      (unsigned)K, GBK, GBN);
    }
    v16h af[2], bf[4];
#pragma unroll
    for (int mi = 0; mi < 2; ++mi)
      af[mi] = load_a_frag(&As[cur][0], wm + mi * 16, GBK, 0);
#pragma unroll
    for (int ni = 0; ni < 4; ++ni)
      bf[ni] = load_b_frag(&Bs[cur][0], wn + ni * 16, GBK, 0);
#pragma unroll
    for (int mi = 0; mi < 2; ++mi)
#pragma unroll
      for (int ni = 0; ni < 4; ++ni)
        acc[mi][ni] = wmma_f16(af[mi], bf[ni], acc[mi][ni]);
  }

  const int lane = tid & 31;
  const int colL = lane & 15;
  const int rofs = (lane & 16) ? 8 : 0;
#pragma unroll
  for (int mi = 0; mi < 2; ++mi)
#pragma unroll
    for (int ni = 0; ni < 4; ++ni) {
      int col = bn + wn + ni * 16 + colL;
      float bv = bias[col];
      float* cp = C + (size_t)(bm + wm + mi * 16 + rofs) * N + col;
#pragma unroll
      for (int v = 0; v < 8; ++v) {
        *cp = acc[mi][ni][v] + bv;
        cp += N;
      }
    }
}

// ---------------------------------------------------------------------------
// One-time fp32 -> f16 conversion (for x), vectorized x4.
// ---------------------------------------------------------------------------
__global__ __launch_bounds__(256) void cvt_f32_to_f16(
    const float* __restrict__ src, _Float16* __restrict__ dst) {
  int i = (blockIdx.x * 256 + threadIdx.x) * 4;
  float4 v = *(const float4*)(src + i);
  _Float16* d = dst + i;
  d[0] = (_Float16)v.x; d[1] = (_Float16)v.y;
  d[2] = (_Float16)v.z; d[3] = (_Float16)v.w;
}

// ---------------------------------------------------------------------------
// One-time weight transpose + f16: W [K,N] fp32 -> Wt [N,K] f16 (32x32 tiles).
// ---------------------------------------------------------------------------
__global__ __launch_bounds__(256) void transpose_to_f16(
    const float* __restrict__ W, _Float16* __restrict__ Wt, int K, int N) {
  __shared__ _Float16 tile[32][33];
  int n0 = blockIdx.x * 32, k0 = blockIdx.y * 32;
  int tx = threadIdx.x & 31, ty = threadIdx.x >> 5;  // 8 rows per pass
#pragma unroll
  for (int p = 0; p < 4; ++p) {
    int k = k0 + ty + p * 8;
    tile[ty + p * 8][tx] = (_Float16)W[(size_t)k * N + n0 + tx];
  }
  __syncthreads();
#pragma unroll
  for (int p = 0; p < 4; ++p) {
    int n = n0 + ty + p * 8;
    Wt[(size_t)n * K + k0 + tx] = tile[tx][ty + p * 8];
  }
}

// ---------------------------------------------------------------------------
// RoPE + split/pack heads: qkv fp32 [B,T,3C] -> Qh/Kh/Vh f16 [B,H,T,D]
// ---------------------------------------------------------------------------
__global__ __launch_bounds__(256) void rope_pack(
    const float* __restrict__ qkv, _Float16* __restrict__ Qh,
    _Float16* __restrict__ Kh, _Float16* __restrict__ Vh) {
  int idx = blockIdx.x * 256 + threadIdx.x;  // 2*16*2048*32 threads
  int i = idx & 31;
  int t = (idx >> 5) & 2047;
  int h = (idx >> 16) & 15;
  int b = (idx >> 20) & 1;

  const float* base = qkv + ((size_t)(b * 2048 + t)) * 3072 + h * 64;
  float inv_freq = __powf(10000.0f, -(float)(2 * i) * (1.0f / 64.0f));
  float s, c;
  __sincosf((float)t * inv_freq, &s, &c);

  size_t o = ((size_t)(b * 16 + h) * 2048 + t) * 64;
  float q1 = base[2 * i], q2 = base[2 * i + 1];
  Qh[o + i]      = (_Float16)(q1 * c - q2 * s);
  Qh[o + 32 + i] = (_Float16)(q1 * s + q2 * c);
  float k1 = base[1024 + 2 * i], k2 = base[1024 + 2 * i + 1];
  Kh[o + i]      = (_Float16)(k1 * c - k2 * s);
  Kh[o + 32 + i] = (_Float16)(k1 * s + k2 * c);
  Vh[o + 2 * i]     = (_Float16)base[2048 + 2 * i];
  Vh[o + 2 * i + 1] = (_Float16)base[2048 + 2 * i + 1];
}

// ---------------------------------------------------------------------------
// Flash attention (causal, online softmax), WMMA for QK^T and PV.
// Block = (bh, 128 q rows); 8 waves each own a 16-row strip; 32-key tiles.
// Output written f16 in [B,T,C] layout for the TDM projection GEMM.
// ---------------------------------------------------------------------------
#define AT_BQ 128
#define AT_BK 32

__global__ __launch_bounds__(256) void attn_flash_wmma(
    const _Float16* __restrict__ Qh, const _Float16* __restrict__ Kh,
    const _Float16* __restrict__ Vh, _Float16* __restrict__ Y) {
  const int T = 2048, D = 64;
  const int bh = blockIdx.y;
  const int q0 = blockIdx.x * AT_BQ;
  const int tid = threadIdx.x;
  const int wid = tid >> 5;
  const int lane = tid & 31;
  const int qbase = q0 + wid * 16;

  const _Float16* Qp = Qh + (size_t)bh * T * D;
  const _Float16* Kp = Kh + (size_t)bh * T * D;
  const _Float16* Vp = Vh + (size_t)bh * T * D;

  __shared__ __align__(16) _Float16 Ks[AT_BK * 64];  // [key][d]
  __shared__ __align__(16) _Float16 Vt[64 * AT_BK];  // [d][key]
  __shared__ __align__(16) _Float16 Ps[8][16 * 32];  // per-wave P [q][key]

  v16h qa0 = load_a_frag(Qp, qbase, D, 0);
  v16h qa1 = load_a_frag(Qp, qbase, D, 32);

  v8f O[4];
#pragma unroll
  for (int ni = 0; ni < 4; ++ni) O[ni] = {};
  float rmax[8], rsum[8];
#pragma unroll
  for (int v = 0; v < 8; ++v) { rmax[v] = -1e30f; rsum[v] = 0.0f; }

  const int rofs = (lane & 16) ? 8 : 0;
  const int colL = lane & 15;
  const int kend = q0 + AT_BQ;

  for (int kt = 0; kt < kend; kt += AT_BK) {
    {
      int r = tid >> 3;
      int c = (tid & 7) * 8;
      *(v8h*)(Ks + r * 64 + c) = *(const v8h*)(Kp + (size_t)(kt + r) * D + c);
      v8h vv = *(const v8h*)(Vp + (size_t)(kt + r) * D + c);
#pragma unroll
      for (int i = 0; i < 8; ++i) Vt[(c + i) * AT_BK + r] = vv[i];
    }
    __syncthreads();

    v8f s[2];
#pragma unroll
    for (int nt = 0; nt < 2; ++nt) {
      v16h b0 = load_b_frag(Ks, nt * 16, 64, 0);
      v16h b1 = load_b_frag(Ks, nt * 16, 64, 32);
      v8f a = {};
      a = wmma_f16(qa0, b0, a);
      a = wmma_f16(qa1, b1, a);
      s[nt] = a;
    }

    float tmax[8];
#pragma unroll
    for (int v = 0; v < 8; ++v) tmax[v] = -1e30f;
#pragma unroll
    for (int nt = 0; nt < 2; ++nt)
#pragma unroll
      for (int v = 0; v < 8; ++v) {
        int colg = kt + nt * 16 + colL;
        int rowg = qbase + rofs + v;
        float val = s[nt][v] * 0.125f;  // 1/sqrt(64)
        if (colg > rowg) val = -1e30f;
        s[nt][v] = val;
        tmax[v] = fmaxf(tmax[v], val);
      }
#pragma unroll
    for (int off = 1; off < 16; off <<= 1)
#pragma unroll
      for (int v = 0; v < 8; ++v)
        tmax[v] = fmaxf(tmax[v], __shfl_xor(tmax[v], off, 32));

    float scale[8];
#pragma unroll
    for (int v = 0; v < 8; ++v) {
      float nm = fmaxf(rmax[v], tmax[v]);
      scale[v] = __expf(rmax[v] - nm);
      rmax[v] = nm;
    }
    float psum[8];
#pragma unroll
    for (int v = 0; v < 8; ++v) psum[v] = 0.0f;
#pragma unroll
    for (int nt = 0; nt < 2; ++nt)
#pragma unroll
      for (int v = 0; v < 8; ++v) {
        float p = __expf(s[nt][v] - rmax[v]);
        psum[v] += p;
        Ps[wid][(rofs + v) * 32 + nt * 16 + colL] = (_Float16)p;
      }
#pragma unroll
    for (int off = 1; off < 16; off <<= 1)
#pragma unroll
      for (int v = 0; v < 8; ++v) psum[v] += __shfl_xor(psum[v], off, 32);
#pragma unroll
    for (int v = 0; v < 8; ++v) rsum[v] = rsum[v] * scale[v] + psum[v];
#pragma unroll
    for (int ni = 0; ni < 4; ++ni)
#pragma unroll
      for (int v = 0; v < 8; ++v) O[ni][v] *= scale[v];

    // DS ops are in-order within a wave: safe to read back our own P tile.
    v16h pa = load_a_frag(&Ps[wid][0], 0, 32, 0);
#pragma unroll
    for (int ni = 0; ni < 4; ++ni) {
      v16h vb = load_b_frag(Vt, ni * 16, AT_BK, 0);
      O[ni] = wmma_f16(pa, vb, O[ni]);
    }
    __syncthreads();
  }

  const int b = bh >> 4, h = bh & 15;
#pragma unroll
  for (int ni = 0; ni < 4; ++ni)
#pragma unroll
    for (int v = 0; v < 8; ++v) {
      int t = qbase + rofs + v;
      int d = ni * 16 + colL;
      Y[((size_t)(b * T + t)) * 1024 + h * 64 + d] =
          (_Float16)(O[ni][v] / rsum[v]);
    }
}

// ---------------------------------------------------------------------------
// Launch: cvt/transpose -> qkv GEMM (TDM) -> rope -> flash attn -> proj GEMM
// ---------------------------------------------------------------------------
extern "C" void kernel_launch(void* const* d_in, const int* in_sizes, int n_in,
                              void* d_out, int out_size, void* d_ws,
                              size_t ws_size, hipStream_t stream) {
  const float* x      = (const float*)d_in[0];
  const float* W_attn = (const float*)d_in[1];
  const float* b_attn = (const float*)d_in[2];
  const float* W_proj = (const float*)d_in[3];
  const float* b_proj = (const float*)d_in[4];
  float* out = (float*)d_out;

  const int BT = 4096, C = 1024;

  // Workspace layout (all chunks 16B-aligned by construction)
  char* p = (char*)d_ws;
  float* qkv = (float*)p;                 p += (size_t)BT * 3 * C * 4;  // 48MB
  _Float16* Xh  = (_Float16*)p;           p += (size_t)BT * C * 2;      //  8MB
  _Float16* Wat = (_Float16*)p;           p += (size_t)3 * C * C * 2;   //  6MB
  _Float16* Wpt = (_Float16*)p;           p += (size_t)C * C * 2;       //  2MB
  _Float16* Yh  = (_Float16*)p;           p += (size_t)BT * C * 2;      //  8MB
  _Float16* Qh  = (_Float16*)p;           p += (size_t)BT * C * 2;
  _Float16* Kh  = (_Float16*)p;           p += (size_t)BT * C * 2;
  _Float16* Vh  = (_Float16*)p;

  // 0) one-time conversions / weight transposes to f16
  cvt_f32_to_f16<<<(BT * C) / (256 * 4), 256, 0, stream>>>(x, Xh);
  transpose_to_f16<<<dim3(3 * C / 32, C / 32), 256, 0, stream>>>(W_attn, Wat,
                                                                 C, 3 * C);
  transpose_to_f16<<<dim3(C / 32, C / 32), 256, 0, stream>>>(W_proj, Wpt, C, C);

  // 1) qkv = x @ W_attn + b_attn   (M=4096, N=3072, K=1024)
  gemm_tdm_wmma<<<dim3(3 * C / GBN, BT / GBM), 256, 0, stream>>>(
      Xh, Wat, b_attn, qkv, BT, 3 * C, C);

  // 2) RoPE + head split/pack
  rope_pack<<<(2 * 16 * 2048 * 32) / 256, 256, 0, stream>>>(qkv, Qh, Kh, Vh);

  // 3) causal flash attention -> Yh (f16, [B,T,C])
  attn_flash_wmma<<<dim3(2048 / AT_BQ, 32), 256, 0, stream>>>(Qh, Kh, Vh, Yh);

  // 4) out = Y @ W_proj + b_proj   (M=4096, N=1024, K=1024)
  gemm_tdm_wmma<<<dim3(C / GBN, BT / GBM), 256, 0, stream>>>(
      Yh, Wpt, b_proj, out, BT, C, C);
}
